// RNN_17669495456222
// MI455X (gfx1250) — matrix operand order repaired and checked by reading, not compiled, not run on hardware
//
#include <hip/hip_runtime.h>
#include <hip/hip_bf16.h>

#define FEATURES   128
#define INPUT_SIZE 1024
#define HIDDEN     1024
#define BATCH      1024
#define SEQ_LEN    192
#define PRED_LEN   48
#define TSTEPS     (SEQ_LEN + PRED_LEN - 1)   // 239

// LDS row padding: +8 bf16 (16B) rotates rows by 4 banks -> conflict-free
// 32-lane ds_load_b128 (each bank touched exactly twice = minimum).
#define PAD   8
#define LDU   (FEATURES + PAD)     // 136
#define LDH   (HIDDEN + PAD)       // 1032

typedef __attribute__((ext_vector_type(16))) __bf16 v16bf;
typedef __attribute__((ext_vector_type(8)))  float  v8f;

union ABFrag { v16bf v; uint4 q[2]; };

// A fragment (16x32 bf16, MxK) from LDS.
// Lanes 0-15: row M=lane, halves K = k0..k0+7 and k0+16..k0+23
// Lanes 16-31: row M=lane-16, halves K = k0+8..k0+15 and k0+24..k0+31
__device__ __forceinline__ v16bf load_a_lds(const __bf16* A, int lda, int k0, int lane) {
  const int m  = lane & 15;
  const int kb = k0 + ((lane & 16) ? 8 : 0);
  const __bf16* p = A + m * lda + kb;
  ABFrag f;
  f.q[0] = *(const uint4*)(p);
  f.q[1] = *(const uint4*)(p + 16);
  return f.v;
}

// B fragment (32x16 bf16, KxN) where B = W^T and W is row-major [N][K]:
// lane's 16 K values for column n are two 16B reads of W row n: K = kc..kc+7 and kc+16..kc+23.
__device__ __forceinline__ v16bf load_b_glb(const __bf16* W, int ldk, int nb, int k0, int lane) {
  const int n  = nb + (lane & 15);
  const int kc = k0 + ((lane & 16) ? 8 : 0);
  const __bf16* p = W + (size_t)n * ldk + kc;
  ABFrag f;
  f.q[0] = *(const uint4*)(p);
  f.q[1] = *(const uint4*)(p + 16);
  __builtin_prefetch((const void*)(p + 32), 0, 1);   // next k-step of this row
  return f.v;
}

// acc[NT] (each 16x16 f32) += A(16xktot) * W^T slice, K-major loop of 32.
template<int NT>
__device__ __forceinline__ void gemm_acc(v8f* acc, const __bf16* A, int lda, int ktot,
                                         const __bf16* W, int nb, int lane) {
#pragma unroll 2
  for (int k0 = 0; k0 < ktot; k0 += 32) {
    v16bf a = load_a_lds(A, lda, k0, lane);
#pragma unroll
    for (int nt = 0; nt < NT; ++nt) {
      v16bf b = load_b_glb(W, ktot, nb + nt * 16, k0, lane);
      acc[nt] = __builtin_amdgcn_wmma_f32_16x16x32_bf16(
          /*neg_a=*/false, a, /*neg_b=*/false, b,
          /*c_mod=*/(short)0, acc[nt], /*reuse_a=*/false, /*reuse_b=*/false);
    }
  }
}

// tanh via v_exp_f32 + v_rcp_f32; bf16-accurate, ~6 VALU ops.
__device__ __forceinline__ float fast_tanh(float x) {
  x = fminf(15.0f, fmaxf(-15.0f, x));
  const float e = __expf(2.0f * x);
  return 1.0f - 2.0f * __builtin_amdgcn_rcpf(e + 1.0f);
}

__global__ void __launch_bounds__(256)
rnn_persistent(const float* __restrict__ x,
               const __bf16* __restrict__ Wenc, const __bf16* __restrict__ Wih0,
               const __bf16* __restrict__ Whh0, const __bf16* __restrict__ Wih1,
               const __bf16* __restrict__ Whh1, const __bf16* __restrict__ Wdec,
               const float* __restrict__ benc, const float* __restrict__ bih0,
               const float* __restrict__ bhh0, const float* __restrict__ bih1,
               const float* __restrict__ bhh1, const float* __restrict__ bdec,
               float* __restrict__ out)
{
  __shared__ __align__(16) __bf16 s_u [16][LDU];
  __shared__ __align__(16) __bf16 s_e [16][LDH];
  __shared__ __align__(16) __bf16 s_h0[16][LDH];
  __shared__ __align__(16) __bf16 s_h1[16][LDH];
  __shared__ __align__(16) float  s_out[16][FEATURES];

  const int tid  = threadIdx.x;
  const int lane = tid & 31;
  const int wave = tid >> 5;
  const int b0   = blockIdx.x * 16;       // this WG's batch slab
  const int ncol = lane & 15;
  const int mb   = (lane & 16) ? 8 : 0;   // C/D row base for this lane
  const int nbw  = wave * 128;            // N-slice for the 1024-wide GEMMs

  // Time-invariant biases -> registers (hoisted out of the 239-step loop).
  float bias_e[8], bias_0[8], bias_1[8];
#pragma unroll
  for (int nt = 0; nt < 8; ++nt) {
    const int n = nbw + nt * 16 + ncol;
    bias_e[nt] = benc[n];
    bias_0[nt] = bih0[n] + bhh0[n];
    bias_1[nt] = bih1[n] + bhh1[n];
  }
  const float bias_d = bdec[wave * 16 + ncol];

  // zero recurrent state (padding included; pad lanes are never read by WMMA)
  for (int i = tid; i < 16 * LDH; i += 256) {
    (&s_h0[0][0])[i] = (__bf16)0.0f;
    (&s_h1[0][0])[i] = (__bf16)0.0f;
  }
  for (int i = tid; i < 16 * FEATURES; i += 256) (&s_out[0][0])[i] = 0.0f;
  __syncthreads();

#pragma unroll 1
  for (int t = 0; t < TSTEPS; ++t) {
    // ---- u_t = x[:,t,:] (t < SEQ_LEN) else previous decoder output ----
    for (int i = tid; i < 16 * FEATURES; i += 256) {
      const int m = i >> 7, f = i & 127;
      const float v = (t < SEQ_LEN)
          ? x[((size_t)(b0 + m) * SEQ_LEN + t) * FEATURES + f]
          : s_out[m][f];
      s_u[m][f] = (__bf16)v;
    }
    __syncthreads();

    // ---- encoder: e = u @ Wenc^T + b_enc   (K=128, each wave owns N-slice 128) ----
    {
      v8f acc[8];
#pragma unroll
      for (int nt = 0; nt < 8; ++nt)
#pragma unroll
        for (int r = 0; r < 8; ++r) acc[nt][r] = bias_e[nt];
      gemm_acc<8>(acc, &s_u[0][0], LDU, FEATURES, Wenc, nbw, lane);
#pragma unroll
      for (int nt = 0; nt < 8; ++nt)
#pragma unroll
        for (int r = 0; r < 8; ++r)
          s_e[mb + r][nbw + nt * 16 + ncol] = (__bf16)acc[nt][r];
    }
    __syncthreads();

    // ---- layer 0: h0 = tanh(e @ Wih0^T + h0 @ Whh0^T + b) ----
    {
      v8f acc[8];
#pragma unroll
      for (int nt = 0; nt < 8; ++nt)
#pragma unroll
        for (int r = 0; r < 8; ++r) acc[nt][r] = bias_0[nt];
      gemm_acc<8>(acc, &s_e[0][0],  LDH, INPUT_SIZE, Wih0, nbw, lane);
      gemm_acc<8>(acc, &s_h0[0][0], LDH, HIDDEN,     Whh0, nbw, lane);
      __syncthreads();   // all waves done reading old s_h0
#pragma unroll
      for (int nt = 0; nt < 8; ++nt)
#pragma unroll
        for (int r = 0; r < 8; ++r)
          s_h0[mb + r][nbw + nt * 16 + ncol] = (__bf16)fast_tanh(acc[nt][r]);
    }
    __syncthreads();

    // ---- layer 1: h1 = tanh(h0 @ Wih1^T + h1 @ Whh1^T + b) ----
    {
      v8f acc[8];
#pragma unroll
      for (int nt = 0; nt < 8; ++nt)
#pragma unroll
        for (int r = 0; r < 8; ++r) acc[nt][r] = bias_1[nt];
      gemm_acc<8>(acc, &s_h0[0][0], LDH, HIDDEN, Wih1, nbw, lane);
      gemm_acc<8>(acc, &s_h1[0][0], LDH, HIDDEN, Whh1, nbw, lane);
      __syncthreads();
#pragma unroll
      for (int nt = 0; nt < 8; ++nt)
#pragma unroll
        for (int r = 0; r < 8; ++r)
          s_h1[mb + r][nbw + nt * 16 + ncol] = (__bf16)fast_tanh(acc[nt][r]);
    }
    __syncthreads();

    // ---- decoder: out = h1 @ Wdec^T + b_dec   (N=128 -> one 16x16 tile per wave) ----
    {
      const int nb = wave * 16;
      v8f acc[1];
#pragma unroll
      for (int r = 0; r < 8; ++r) acc[0][r] = bias_d;
      gemm_acc<1>(acc, &s_h1[0][0], LDH, HIDDEN, Wdec, nb, lane);
#pragma unroll
      for (int r = 0; r < 8; ++r)
        s_out[mb + r][nb + ncol] = acc[0][r];
      if (t >= SEQ_LEN - 1) {
        const int p = t - (SEQ_LEN - 1);
#pragma unroll
        for (int r = 0; r < 8; ++r)
          out[((size_t)(b0 + mb + r) * PRED_LEN + p) * FEATURES + nb + ncol] = acc[0][r];
      }
    }
    __syncthreads();
  }
}

__global__ void cvt_bf16_kernel(const float* __restrict__ s, __bf16* __restrict__ d, int n) {
  const int i = blockIdx.x * 256 + threadIdx.x;
  if (i < n) d[i] = (__bf16)s[i];
}

extern "C" void kernel_launch(void* const* d_in, const int* in_sizes, int n_in,
                              void* d_out, int out_size, void* d_ws, size_t ws_size,
                              hipStream_t stream) {
  const float* x     = (const float*)d_in[0];
  const float* Wencf = (const float*)d_in[1];
  const float* benc  = (const float*)d_in[2];
  const float* Wih0f = (const float*)d_in[3];
  const float* bih0  = (const float*)d_in[4];
  const float* Whh0f = (const float*)d_in[5];
  const float* bhh0  = (const float*)d_in[6];
  const float* Wih1f = (const float*)d_in[7];
  const float* bih1  = (const float*)d_in[8];
  const float* Whh1f = (const float*)d_in[9];
  const float* bhh1  = (const float*)d_in[10];
  const float* Wdecf = (const float*)d_in[11];
  const float* bdec  = (const float*)d_in[12];

  __bf16* ws = (__bf16*)d_ws;
  const size_t nEnc = (size_t)INPUT_SIZE * FEATURES;   // 1024x128
  const size_t nHid = (size_t)HIDDEN * HIDDEN;         // 1024x1024
  const size_t nDec = (size_t)FEATURES * HIDDEN;       // 128x1024
  __bf16* Wenc = ws;
  __bf16* Wih0 = Wenc + nEnc;
  __bf16* Whh0 = Wih0 + nHid;
  __bf16* Wih1 = Whh0 + nHid;
  __bf16* Whh1 = Wih1 + nHid;
  __bf16* Wdec = Whh1 + nHid;

  cvt_bf16_kernel<<<(int)((nEnc + 255) / 256), 256, 0, stream>>>(Wencf, Wenc, (int)nEnc);
  cvt_bf16_kernel<<<(int)((nHid + 255) / 256), 256, 0, stream>>>(Wih0f, Wih0, (int)nHid);
  cvt_bf16_kernel<<<(int)((nHid + 255) / 256), 256, 0, stream>>>(Whh0f, Whh0, (int)nHid);
  cvt_bf16_kernel<<<(int)((nHid + 255) / 256), 256, 0, stream>>>(Wih1f, Wih1, (int)nHid);
  cvt_bf16_kernel<<<(int)((nHid + 255) / 256), 256, 0, stream>>>(Whh1f, Whh1, (int)nHid);
  cvt_bf16_kernel<<<(int)((nDec + 255) / 256), 256, 0, stream>>>(Wdecf, Wdec, (int)nDec);

  rnn_persistent<<<BATCH / 16, 256, 0, stream>>>(
      x, Wenc, Wih0, Whh0, Wih1, Whh1, Wdec,
      benc, bih0, bhh0, bih1, bhh1, bdec, (float*)d_out);
}
